// EquivariantTransformerLayer_7275674599758
// MI455X (gfx1250) — matrix operand order repaired and checked by reading, not compiled, not run on hardware
//
#include <hip/hip_runtime.h>
#include <hip/hip_bf16.h>

// MI455X (gfx1250, wave32) implementation.
// Strategy: bf16 WMMA (v_wmma_f32_16x16x32_bf16) for all GEMMs; softmax is
// over the HEADS axis (reference quirk) -> stage 16-head score tiles in LDS,
// softmax elementwise, P@V with pre-transposed V. rel_pos mean is the only
// HBM-bound step (201 MB).

typedef __bf16 bf16_t;
typedef __attribute__((ext_vector_type(16))) __bf16 v16bf;
typedef __attribute__((ext_vector_type(8)))  __bf16 v8bf;
typedef __attribute__((ext_vector_type(8)))  float  v8f;

#define DEVINL __device__ __forceinline__

DEVINL v16bf make_frag(v8bf lo, v8bf hi) {
  v16bf r;
#pragma unroll
  for (int i = 0; i < 8; ++i) { r[i] = lo[i]; r[i + 8] = hi[i]; }
  return r;
}

// A-operand: 16(M) x 32(K) bf16 from row-major [M][K] tile at base, leading dim ld.
// Layout (ISA 7.12.2): lane L -> M=L&15, elems 0..7 = K[(L>>4)*8 .. +7], 8..15 = +16.
DEVINL v16bf load_a_frag(const bf16_t* __restrict__ base, int ld, int lane) {
  int r = lane & 15, half = lane >> 4;
  const bf16_t* p = base + (size_t)r * ld + half * 8;
  v8bf lo = *(const v8bf*)(p);
  v8bf hi = *(const v8bf*)(p + 16);
  return make_frag(lo, hi);
}

// B-operand: 32(K) x 16(N) bf16 taken from B^T stored row-major [N][K].
// Layout: lane L -> N=L&15, elems 0..7 = K[(L>>4)*8 .. +7], 8..15 = +16.
DEVINL v16bf load_b_frag(const bf16_t* __restrict__ baseT, int ld, int lane) {
  int n = lane & 15, half = lane >> 4;
  const bf16_t* p = baseT + (size_t)n * ld + half * 8;
  v8bf lo = *(const v8bf*)(p);
  v8bf hi = *(const v8bf*)(p + 16);
  return make_frag(lo, hi);
}

DEVINL v8f wmma_bf16(v16bf a, v16bf b, v8f c) {
  return __builtin_amdgcn_wmma_f32_16x16x32_bf16(false, a, false, b, (short)0, c,
                                                 false, false);
}

// ---------------------------------------------------------------- prep kernels

__global__ void k_cvt_bf16(const float* __restrict__ in, bf16_t* __restrict__ out,
                           int n) {
  for (int i = blockIdx.x * blockDim.x + threadIdx.x; i < n;
       i += gridDim.x * blockDim.x)
    out[i] = (bf16_t)in[i];
}

// in: f32 [K][N] (ldin = N); out: bf16 [N][K] (ldout = K). Dims multiples of 32.
__global__ __launch_bounds__(256) void k_transpose_f32_bf16(
    const float* __restrict__ in, int ldin, bf16_t* __restrict__ out, int ldout) {
  __shared__ float tile[32][33];
  int tid = threadIdx.x;
  int c = tid & 31, r = tid >> 5;  // 8 rows per pass
  int kbase = blockIdx.y * 32, nbase = blockIdx.x * 32;
#pragma unroll
  for (int p = 0; p < 4; ++p)
    tile[r + p * 8][c] = in[(size_t)(kbase + r + p * 8) * ldin + nbase + c];
  __syncthreads();
#pragma unroll
  for (int p = 0; p < 4; ++p)
    out[(size_t)(nbase + r + p * 8) * ldout + kbase + c] =
        (bf16_t)tile[c][r + p * 8];
}

// ------------------------------------------------------------------- QKV GEMM
// C[4096][3072] = h_bf @ [Wq|Wk|Wv] + bias. Cols 0..2047 (Q,K) -> qk row-major;
// cols 2048..3071 (V) written transposed into Vt[1024][4096].
__global__ __launch_bounds__(256) void k_qkv_gemm(
    const bf16_t* __restrict__ A, const bf16_t* __restrict__ BT,
    const float* __restrict__ bq, const float* __restrict__ bk,
    const float* __restrict__ bv, bf16_t* __restrict__ qk,
    bf16_t* __restrict__ Vt) {
  int tid = threadIdx.x, lane = tid & 31, w = tid >> 5;
  int rw = w & 3, cwv = w >> 2;                 // 4 row-waves x 2 col-waves
  int row0 = blockIdx.x * 128 + rw * 32;
  int col0 = blockIdx.y * 64 + cwv * 32;
  v8f acc[2][2] = {};
  for (int k0 = 0; k0 < 1024; k0 += 32) {
    v16bf a0 = load_a_frag(A + (size_t)row0 * 1024 + k0, 1024, lane);
    v16bf a1 = load_a_frag(A + (size_t)(row0 + 16) * 1024 + k0, 1024, lane);
    v16bf b0 = load_b_frag(BT + (size_t)col0 * 1024 + k0, 1024, lane);
    v16bf b1 = load_b_frag(BT + (size_t)(col0 + 16) * 1024 + k0, 1024, lane);
    acc[0][0] = wmma_bf16(a0, b0, acc[0][0]);
    acc[0][1] = wmma_bf16(a0, b1, acc[0][1]);
    acc[1][0] = wmma_bf16(a1, b0, acc[1][0]);
    acc[1][1] = wmma_bf16(a1, b1, acc[1][1]);
  }
  int lr = lane & 15, lh = lane >> 4;
#pragma unroll
  for (int rs = 0; rs < 2; ++rs)
#pragma unroll
    for (int cs = 0; cs < 2; ++cs) {
      int c = col0 + cs * 16 + lr;
      float bias = (c < 1024) ? bq[c] : (c < 2048 ? bk[c - 1024] : bv[c - 2048]);
#pragma unroll
      for (int i = 0; i < 8; ++i) {
        int rrow = row0 + rs * 16 + i + lh * 8;
        float v = acc[rs][cs][i] + bias;
        if (c < 2048) qk[(size_t)rrow * 2048 + c] = (bf16_t)v;
        else          Vt[(size_t)(c - 2048) * 4096 + rrow] = (bf16_t)v;
      }
    }
}

// ------------------------------------------------------------------ attention
// Per block: 16 query rows. Loop m in chunks of 32:
//  ph1: all waves compute S[n][m][h] (wave w -> heads 2w,2w+1) into LDS
//  ph2: softmax over h (the reference softmaxes over the heads axis!)
//  ph3: P@V accumulated in registers (V pre-transposed: Vt[h*64+d][m]).
__global__ __launch_bounds__(256) void k_attn(const bf16_t* __restrict__ qk,
                                              const bf16_t* __restrict__ Vt,
                                              bf16_t* __restrict__ attn_out) {
  __shared__ float  S[16][32][16];    // [n][m][h]  32 KB
  __shared__ bf16_t P[16][16][32];    // [h][n][m]  16 KB
  int tid = threadIdx.x, lane = tid & 31, w = tid >> 5;
  int n0 = blockIdx.x * 16;
  int lr = lane & 15, lh = lane >> 4;

  v16bf qf[2][2];
#pragma unroll
  for (int hh = 0; hh < 2; ++hh) {
    int h = w * 2 + hh;
#pragma unroll
    for (int ks = 0; ks < 2; ++ks)
      qf[hh][ks] =
          load_a_frag(qk + (size_t)n0 * 2048 + h * 64 + ks * 32, 2048, lane);
  }
  v8f oacc[2][4] = {};

  for (int m0 = 0; m0 < 4096; m0 += 32) {
    // phase 1: scores for this wave's 2 heads
#pragma unroll
    for (int hh = 0; hh < 2; ++hh) {
      int h = w * 2 + hh;
#pragma unroll
      for (int ms = 0; ms < 2; ++ms) {
        v8f sacc = {};
#pragma unroll
        for (int ks = 0; ks < 2; ++ks) {
          v16bf kf = load_b_frag(
              qk + (size_t)(m0 + ms * 16) * 2048 + 1024 + h * 64 + ks * 32, 2048,
              lane);
          sacc = wmma_bf16(qf[hh][ks], kf, sacc);
        }
#pragma unroll
        for (int i = 0; i < 8; ++i)
          S[i + lh * 8][ms * 16 + lr][h] = sacc[i] * 0.125f;  // 1/sqrt(64)
      }
    }
    __syncthreads();
    // phase 2: softmax over the 16 heads, per (n,m) pair
#pragma unroll
    for (int p = 0; p < 2; ++p) {
      int pair = tid + p * 256;  // 512 pairs = 16n x 32m
      int nn = pair >> 5, mm = pair & 31;
      float s[16], mx = -1e30f;
#pragma unroll
      for (int h = 0; h < 16; ++h) { s[h] = S[nn][mm][h]; mx = fmaxf(mx, s[h]); }
      float sum = 0.f;
#pragma unroll
      for (int h = 0; h < 16; ++h) { s[h] = __expf(s[h] - mx); sum += s[h]; }
      float inv = 1.f / sum;
#pragma unroll
      for (int h = 0; h < 16; ++h) P[h][nn][mm] = (bf16_t)(s[h] * inv);
    }
    __syncthreads();
    // phase 3: out[n][h][d] += P[n][m][h] * V[m][h][d]
#pragma unroll
    for (int hh = 0; hh < 2; ++hh) {
      int h = w * 2 + hh;
      const bf16_t* pa = &P[h][0][0] + lr * 32 + lh * 8;  // A-frag from LDS
      v8bf lo = *(const v8bf*)(pa);
      v8bf hi = *(const v8bf*)(pa + 16);
      v16bf af = make_frag(lo, hi);
#pragma unroll
      for (int ds = 0; ds < 4; ++ds) {
        v16bf vf =
            load_b_frag(Vt + (size_t)(h * 64 + ds * 16) * 4096 + m0, 4096, lane);
        oacc[hh][ds] = wmma_bf16(af, vf, oacc[hh][ds]);
      }
    }
    __syncthreads();
  }
#pragma unroll
  for (int hh = 0; hh < 2; ++hh) {
    int h = w * 2 + hh;
#pragma unroll
    for (int ds = 0; ds < 4; ++ds)
#pragma unroll
      for (int i = 0; i < 8; ++i) {
        int n = n0 + i + lh * 8;
        attn_out[(size_t)n * 1024 + h * 64 + ds * 16 + lr] =
            (bf16_t)oacc[hh][ds][i];
      }
  }
}

// --------------------------------------------- out-proj + residual + LayerNorm
__global__ __launch_bounds__(256) void k_oproj_ln(
    const bf16_t* __restrict__ A, const bf16_t* __restrict__ BT,
    const float* __restrict__ bo, const float* __restrict__ hres,
    const float* __restrict__ lnw, const float* __restrict__ lnb,
    float* __restrict__ hn_out, bf16_t* __restrict__ hn_bf) {
  __shared__ float rsum[16], rsq[16], mu_s[16], rstd_s[16];
  int tid = threadIdx.x, lane = tid & 31, w = tid >> 5;
  int n0 = blockIdx.x * 16;
  int lr = lane & 15, lh = lane >> 4;
  if (tid < 16) { rsum[tid] = 0.f; rsq[tid] = 0.f; }
  __syncthreads();
  v8f acc[8] = {};
  for (int k0 = 0; k0 < 1024; k0 += 32) {
    v16bf a = load_a_frag(A + (size_t)n0 * 1024 + k0, 1024, lane);
#pragma unroll
    for (int t = 0; t < 8; ++t) {
      v16bf b =
          load_b_frag(BT + (size_t)(w * 128 + t * 16) * 1024 + k0, 1024, lane);
      acc[t] = wmma_bf16(a, b, acc[t]);
    }
  }
  float x[8][8];
#pragma unroll
  for (int t = 0; t < 8; ++t) {
    int c = w * 128 + t * 16 + lr;
    float bias = bo[c];
#pragma unroll
    for (int i = 0; i < 8; ++i) {
      int n = n0 + i + lh * 8;
      x[t][i] = acc[t][i] + bias + hres[(size_t)n * 1024 + c];
    }
  }
#pragma unroll
  for (int i = 0; i < 8; ++i) {
    float s = 0.f, q = 0.f;
#pragma unroll
    for (int t = 0; t < 8; ++t) { s += x[t][i]; q += x[t][i] * x[t][i]; }
#pragma unroll
    for (int m = 1; m <= 8; m <<= 1) {
      s += __shfl_xor(s, m, 32);
      q += __shfl_xor(q, m, 32);
    }
    if (lr == 0) { atomicAdd(&rsum[i + lh * 8], s); atomicAdd(&rsq[i + lh * 8], q); }
  }
  __syncthreads();
  if (tid < 16) {
    float mu = rsum[tid] * (1.f / 1024.f);
    float var = rsq[tid] * (1.f / 1024.f) - mu * mu;
    mu_s[tid] = mu;
    rstd_s[tid] = rsqrtf(var + 1e-5f);
  }
  __syncthreads();
#pragma unroll
  for (int t = 0; t < 8; ++t) {
    int c = w * 128 + t * 16 + lr;
    float g = lnw[c], bb = lnb[c];
#pragma unroll
    for (int i = 0; i < 8; ++i) {
      int r = i + lh * 8;
      int n = n0 + r;
      float y = (x[t][i] - mu_s[r]) * rstd_s[r] * g + bb;
      hn_out[(size_t)n * 1024 + c] = y;
      hn_bf[(size_t)n * 1024 + c] = (bf16_t)y;
    }
  }
}

// ------------------------------------------------- coord MLP -> scalar per row
__global__ __launch_bounds__(256) void k_coord_mlp(
    const bf16_t* __restrict__ A, const bf16_t* __restrict__ BT,
    const float* __restrict__ b1, const float* __restrict__ W2,
    const float* __restrict__ b2, float* __restrict__ cw) {
  __shared__ float rsum[16];
  int tid = threadIdx.x, lane = tid & 31, w = tid >> 5;
  int n0 = blockIdx.x * 16;
  int lr = lane & 15, lh = lane >> 4;
  if (tid < 16) rsum[tid] = 0.f;
  __syncthreads();
  v8f acc[4] = {};
  for (int k0 = 0; k0 < 1024; k0 += 32) {
    v16bf a = load_a_frag(A + (size_t)n0 * 1024 + k0, 1024, lane);
#pragma unroll
    for (int t = 0; t < 4; ++t) {
      v16bf b =
          load_b_frag(BT + (size_t)(w * 64 + t * 16) * 1024 + k0, 1024, lane);
      acc[t] = wmma_bf16(a, b, acc[t]);
    }
  }
  float part[8];
#pragma unroll
  for (int i = 0; i < 8; ++i) part[i] = 0.f;
#pragma unroll
  for (int t = 0; t < 4; ++t) {
    int c = w * 64 + t * 16 + lr;
    float bb = b1[c], w2 = W2[c];
#pragma unroll
    for (int i = 0; i < 8; ++i) {
      float v = acc[t][i] + bb;
      float sl = v / (1.f + __expf(-v));  // SiLU
      part[i] += sl * w2;
    }
  }
#pragma unroll
  for (int i = 0; i < 8; ++i) {
    float s = part[i];
#pragma unroll
    for (int m = 1; m <= 8; m <<= 1) s += __shfl_xor(s, m, 32);
    if (lr == 0) atomicAdd(&rsum[i + lh * 8], s);
  }
  __syncthreads();
  if (tid < 16) cw[n0 + tid] = rsum[tid] + b2[0];
}

// ------------------------------- pos_update = cw[n] * mean_m rel_pos[n][m][:]
__global__ __launch_bounds__(256) void k_pos_update(
    const float* __restrict__ rel_pos, const float* __restrict__ cw,
    float* __restrict__ pos_out) {
  __shared__ float sums[3];
  int tid = threadIdx.x;
  int n = blockIdx.x;
  if (tid < 3) sums[tid] = 0.f;
  __syncthreads();
  const float* base = rel_pos + (size_t)n * 4096 * 3;
  float s0 = 0.f, s1 = 0.f, s2 = 0.f;
  for (int i = tid; i < 4096 * 3; i += 256) {
    float v = base[i];
    int c = i % 3;
    if (c == 0) s0 += v;
    else if (c == 1) s1 += v;
    else s2 += v;
  }
#pragma unroll
  for (int m = 1; m < 32; m <<= 1) {
    s0 += __shfl_xor(s0, m, 32);
    s1 += __shfl_xor(s1, m, 32);
    s2 += __shfl_xor(s2, m, 32);
  }
  if ((tid & 31) == 0) {
    atomicAdd(&sums[0], s0);
    atomicAdd(&sums[1], s1);
    atomicAdd(&sums[2], s2);
  }
  __syncthreads();
  if (tid < 3)
    pos_out[(size_t)n * 3 + tid] = cw[n] * sums[tid] * (1.0f / 4096.f);
}

// -------------------------------------------------------------------- launcher
extern "C" void kernel_launch(void* const* d_in, const int* in_sizes, int n_in,
                              void* d_out, int out_size, void* d_ws,
                              size_t ws_size, hipStream_t stream) {
  const float* h   = (const float*)d_in[0];
  const float* rel = (const float*)d_in[1];
  const float* Wq  = (const float*)d_in[2];
  const float* bq  = (const float*)d_in[3];
  const float* Wk  = (const float*)d_in[4];
  const float* bk  = (const float*)d_in[5];
  const float* Wv  = (const float*)d_in[6];
  const float* bv  = (const float*)d_in[7];
  const float* Wo  = (const float*)d_in[8];
  const float* bo  = (const float*)d_in[9];
  const float* W1  = (const float*)d_in[10];
  const float* b1  = (const float*)d_in[11];
  const float* W2  = (const float*)d_in[12];
  const float* b2  = (const float*)d_in[13];
  const float* lnw = (const float*)d_in[14];
  const float* lnb = (const float*)d_in[15];
  float* out_hn  = (float*)d_out;
  float* out_pos = out_hn + (size_t)4096 * 1024;

  char* ws = (char*)d_ws;  // ~60 MB used
  bf16_t* h_bf  = (bf16_t*)(ws);
  bf16_t* WqkvT = (bf16_t*)(ws + 8388608);   // [3072][1024]
  bf16_t* WoT   = (bf16_t*)(ws + 14680064);  // [1024][1024]
  bf16_t* W1T   = (bf16_t*)(ws + 16777216);  // [512][1024]
  bf16_t* qk    = (bf16_t*)(ws + 17825792);  // [4096][2048] (Q | K)
  bf16_t* Vt    = (bf16_t*)(ws + 34603008);  // [1024][4096]
  bf16_t* attn  = (bf16_t*)(ws + 42991616);  // [4096][1024]
  bf16_t* hn_bf = (bf16_t*)(ws + 51380224);  // [4096][1024]
  float*  cw    = (float*)(ws + 59768832);   // [4096]

  k_cvt_bf16<<<4096, 256, 0, stream>>>(h, h_bf, 4096 * 1024);
  dim3 tg(1024 / 32, 1024 / 32);
  k_transpose_f32_bf16<<<tg, 256, 0, stream>>>(Wq, 1024, WqkvT, 1024);
  k_transpose_f32_bf16<<<tg, 256, 0, stream>>>(Wk, 1024,
                                               WqkvT + (size_t)1024 * 1024, 1024);
  k_transpose_f32_bf16<<<tg, 256, 0, stream>>>(Wv, 1024,
                                               WqkvT + (size_t)2048 * 1024, 1024);
  k_transpose_f32_bf16<<<tg, 256, 0, stream>>>(Wo, 1024, WoT, 1024);
  dim3 tg1(512 / 32, 1024 / 32);
  k_transpose_f32_bf16<<<tg1, 256, 0, stream>>>(W1, 512, W1T, 1024);

  k_qkv_gemm<<<dim3(4096 / 128, 3072 / 64), 256, 0, stream>>>(h_bf, WqkvT, bq, bk,
                                                              bv, qk, Vt);
  k_attn<<<4096 / 16, 256, 0, stream>>>(qk, Vt, attn);
  k_oproj_ln<<<4096 / 16, 256, 0, stream>>>(attn, WoT, bo, h, lnw, lnb, out_hn,
                                            hn_bf);
  k_coord_mlp<<<4096 / 16, 256, 0, stream>>>(hn_bf, W1T, b1, W2, b2, cw);
  k_pos_update<<<4096, 256, 0, stream>>>(rel, cw, out_pos);
}
